// LSTM_ED_Model_34359738952
// MI455X (gfx1250) — matrix operand order repaired and checked by reading, not compiled, not run on hardware
//
#include <hip/hip_runtime.h>
#include <hip/hip_bf16.h>

// ---------------- CDNA5 WMMA types ----------------
typedef __attribute__((ext_vector_type(16))) _Float16 v16h;
typedef __attribute__((ext_vector_type(8)))  float    v8f;

#define Bdim 8
#define Tdim 64
#define Ndim 512
#define Hdim 256
#define OUTdim 512
#define FLEN 24

__device__ __forceinline__ float sigmoidf_(float x) {
    return 1.0f / (1.0f + __expf(-x));
}

// A-fragment for v_wmma_f32_16x16x32_f16 (f32 source, converted on the fly).
// Lane L (L%16 = row M). K index of element e: (L/16)*16 + e.
__device__ __forceinline__ v16h load_A(const float* __restrict__ X, int stride,
                                       int lm, int half, int k0, int mvalid) {
    v16h a;
    if (lm < mvalid) {
        const float* row = X + lm * stride;
        int kbase = k0 + half * 16;
#pragma unroll
        for (int e = 0; e < 8; ++e) a[e]     = (_Float16)row[kbase + e];
#pragma unroll
        for (int e = 0; e < 8; ++e) a[8 + e] = (_Float16)row[kbase + 8 + e];
    } else {
#pragma unroll
        for (int e = 0; e < 16; ++e) a[e] = (_Float16)0.0f;
    }
    return a;
}

// B-fragment from PRE-PACKED f16 weights (row-major [rows][strideW] halves).
// Lane L holds column N = L%16; K of element e: (L/16)*16 + e  -> 16 contiguous
// halves = 32 bytes = two b128 loads, no conversion in the hot loop.
__device__ __forceinline__ v16h load_Bh(const _Float16* __restrict__ W, int strideW,
                                        int nrow, int half, int k0) {
    const _Float16* p = W + (long)nrow * strideW + k0 + half * 16;
    return *(const v16h*)p;   // 32B-aligned by construction
}

// ---------------- init: zero a float region ----------------
__global__ void init_zero_kernel(float* __restrict__ p, int n) {
    int i = blockIdx.x * blockDim.x + threadIdx.x;
    if (i < n) p[i] = 0.0f;
}

// ---------------- pack: f32 weights -> f16 copy (done once per call) -------
__global__ void pack_f16_kernel(const float* __restrict__ src,
                                _Float16* __restrict__ dst, int n) {
    int i = blockIdx.x * blockDim.x + threadIdx.x;
    if (i < n) dst[i] = (_Float16)src[i];
}

// ---------------- GAT: one block per (b,t) ----------------
__global__ __launch_bounds__(256)
void gat_kernel(const float* __restrict__ x, const float* __restrict__ adj,
                const float* __restrict__ gat_W, const float* __restrict__ gat_a,
                float* __restrict__ xg) {
    const int bt = blockIdx.x;                  // 0 .. B*T-1
    const float* xr = x + (long)bt * Ndim;
    __shared__ float f[Ndim];
    const float W  = gat_W[0];
    const float a0 = gat_a[0];
    const float a1 = gat_a[1];
    for (int j = threadIdx.x; j < Ndim; j += blockDim.x) f[j] = xr[j] * W;
    __syncthreads();

    for (int i = threadIdx.x; i < Ndim; i += blockDim.x) {
        const float ci = a0 * f[i];
        const float* arow = adj + (long)i * Ndim;
        // pass 1: row max
        float mx = -3.0e38f;
        for (int j = 0; j < Ndim; ++j) {
            float v = ci + a1 * f[j];
            v = (v > 0.0f) ? v : 0.2f * v;              // leaky_relu 0.2
            v = (arow[j] > 0.0f) ? v : -9.0e15f;        // adj mask
            mx = fmaxf(mx, v);
        }
        // pass 2: softmax-weighted sum of f
        float s = 0.0f, w = 0.0f;
        for (int j = 0; j < Ndim; ++j) {
            float v = ci + a1 * f[j];
            v = (v > 0.0f) ? v : 0.2f * v;
            v = (arow[j] > 0.0f) ? v : -9.0e15f;
            float e = __expf(v - mx);
            s += e;
            w += e * f[j];
        }
        float r = w / s;
        xg[(long)bt * Ndim + i] = fmaxf(r, 0.0f);       // relu
    }
}

// ---------------- LSTM step: 16 blocks x 1 wave; 4 gate tiles per block ----
// G[8,1024] = X[8,Kx] @ Wih^T + Hin[8,256] @ Whh^T + (bih+bhh); gates -> h,c
__global__ __launch_bounds__(32)
void lstm_step_kernel(const float* __restrict__ X, int Kx, int strideX,
                      const _Float16* __restrict__ Wih,   // packed f16 [1024,Kx]
                      const _Float16* __restrict__ Whh,   // packed f16 [1024,256]
                      const float* __restrict__ bih,
                      const float* __restrict__ bhh,
                      const float* __restrict__ hin,
                      const float* __restrict__ cin,
                      float* __restrict__ hout,
                      float* __restrict__ cout) {
    const int lane = threadIdx.x;
    const int half = lane >> 4;
    const int lm   = lane & 15;
    const int j0   = blockIdx.x * 16;           // hidden-unit tile [j0, j0+16)

    v8f acc_i = {}, acc_f = {}, acc_g = {}, acc_o = {};

    // ---- X @ Wih^T (K = Kx) ----
    for (int k0 = 0; k0 < Kx; k0 += 32) {
        v16h a  = load_A(X, strideX, lm, half, k0, Bdim);
        v16h bi = load_Bh(Wih, Kx, 0 * Hdim + j0 + lm, half, k0);
        v16h bf = load_Bh(Wih, Kx, 1 * Hdim + j0 + lm, half, k0);
        v16h bg = load_Bh(Wih, Kx, 2 * Hdim + j0 + lm, half, k0);
        v16h bo = load_Bh(Wih, Kx, 3 * Hdim + j0 + lm, half, k0);
        acc_i = __builtin_amdgcn_wmma_f32_16x16x32_f16(false, a, false, bi, (short)0, acc_i, false, false);
        acc_f = __builtin_amdgcn_wmma_f32_16x16x32_f16(false, a, false, bf, (short)0, acc_f, false, false);
        acc_g = __builtin_amdgcn_wmma_f32_16x16x32_f16(false, a, false, bg, (short)0, acc_g, false, false);
        acc_o = __builtin_amdgcn_wmma_f32_16x16x32_f16(false, a, false, bo, (short)0, acc_o, false, false);
    }
    // ---- Hin @ Whh^T (K = Hdim) ----
    for (int k0 = 0; k0 < Hdim; k0 += 32) {
        v16h a  = load_A(hin, Hdim, lm, half, k0, Bdim);
        v16h bi = load_Bh(Whh, Hdim, 0 * Hdim + j0 + lm, half, k0);
        v16h bf = load_Bh(Whh, Hdim, 1 * Hdim + j0 + lm, half, k0);
        v16h bg = load_Bh(Whh, Hdim, 2 * Hdim + j0 + lm, half, k0);
        v16h bo = load_Bh(Whh, Hdim, 3 * Hdim + j0 + lm, half, k0);
        acc_i = __builtin_amdgcn_wmma_f32_16x16x32_f16(false, a, false, bi, (short)0, acc_i, false, false);
        acc_f = __builtin_amdgcn_wmma_f32_16x16x32_f16(false, a, false, bf, (short)0, acc_f, false, false);
        acc_g = __builtin_amdgcn_wmma_f32_16x16x32_f16(false, a, false, bg, (short)0, acc_g, false, false);
        acc_o = __builtin_amdgcn_wmma_f32_16x16x32_f16(false, a, false, bo, (short)0, acc_o, false, false);
    }

    // D layout: VGPR r, lanes 0-15 -> M=r, N=lm; lanes 16-31 -> M=8+r
    const int j = j0 + lm;
#pragma unroll
    for (int r = 0; r < 8; ++r) {
        int m = r + half * 8;
        if (m < Bdim) {
            float gi = acc_i[r] + bih[0 * Hdim + j] + bhh[0 * Hdim + j];
            float gf = acc_f[r] + bih[1 * Hdim + j] + bhh[1 * Hdim + j];
            float gg = acc_g[r] + bih[2 * Hdim + j] + bhh[2 * Hdim + j];
            float go = acc_o[r] + bih[3 * Hdim + j] + bhh[3 * Hdim + j];
            float cold = cin[m * Hdim + j];
            float cn = sigmoidf_(gf) * cold + sigmoidf_(gi) * tanhf(gg);
            float hn = sigmoidf_(go) * tanhf(cn);
            cout[m * Hdim + j] = cn;
            hout[m * Hdim + j] = hn;
        }
    }
}

// ---------------- FC head: pred = h @ fc_W^T + fc_b; 32 blocks x 1 wave ----
__global__ __launch_bounds__(32)
void fc_kernel(const float* __restrict__ h,
               const _Float16* __restrict__ fc_Wh,   // packed f16 [512,256]
               const float* __restrict__ fc_b,
               float* __restrict__ dinp,
               float* __restrict__ out, int step) {
    const int lane = threadIdx.x;
    const int half = lane >> 4;
    const int lm   = lane & 15;
    const int n0   = blockIdx.x * 16;

    v8f acc = {};
    for (int k0 = 0; k0 < Hdim; k0 += 32) {
        v16h a = load_A(h, Hdim, lm, half, k0, Bdim);
        v16h b = load_Bh(fc_Wh, Hdim, n0 + lm, half, k0);
        acc = __builtin_amdgcn_wmma_f32_16x16x32_f16(false, a, false, b, (short)0, acc, false, false);
    }
    const int n = n0 + lm;
    const float bias = fc_b[n];
#pragma unroll
    for (int r = 0; r < 8; ++r) {
        int m = r + half * 8;
        if (m < Bdim) {
            float p = acc[r] + bias;
            out[(long)m * FLEN * OUTdim + (long)step * OUTdim + n] = p;
            dinp[m * OUTdim + n] = p;
        }
    }
}

// ---------------- host-side orchestration ----------------
extern "C" void kernel_launch(void* const* d_in, const int* in_sizes, int n_in,
                              void* d_out, int out_size, void* d_ws, size_t ws_size,
                              hipStream_t stream) {
    const float* x        = (const float*)d_in[0];
    const float* adj      = (const float*)d_in[1];
    const float* gat_W    = (const float*)d_in[2];
    const float* gat_a    = (const float*)d_in[3];
    const float* enc_Wih  = (const float*)d_in[4];
    const float* enc_Whh  = (const float*)d_in[5];
    const float* enc_bih  = (const float*)d_in[6];
    const float* enc_bhh  = (const float*)d_in[7];
    const float* dec_Wih  = (const float*)d_in[8];
    const float* dec_Whh  = (const float*)d_in[9];
    const float* dec_bih  = (const float*)d_in[10];
    const float* dec_bhh  = (const float*)d_in[11];
    const float* fc_W     = (const float*)d_in[12];
    const float* fc_b     = (const float*)d_in[13];
    float* out = (float*)d_out;
    float* ws  = (float*)d_ws;

    // ---- workspace layout (float offsets; all 256-float aligned) ----
    float* xg   = ws;                               // B*T*N   = 262144 floats
    float* h0   = xg + (long)Bdim * Tdim * Ndim;    // 2048
    float* h1   = h0 + Bdim * Hdim;
    float* c0   = h1 + Bdim * Hdim;
    float* c1   = c0 + Bdim * Hdim;
    float* dinp = c1 + Bdim * Hdim;                 // B*OUT = 4096
    // packed f16 weights (resident in L2 across the whole recurrence)
    _Float16* wh = (_Float16*)(dinp + Bdim * OUTdim);
    _Float16* enc_Wih_h = wh;                                   // 1024*512
    _Float16* enc_Whh_h = enc_Wih_h + 1024 * Ndim;              // 1024*256
    _Float16* dec_Wih_h = enc_Whh_h + 1024 * Hdim;              // 1024*512
    _Float16* dec_Whh_h = dec_Wih_h + 1024 * OUTdim;            // 1024*256
    _Float16* fc_W_h    = dec_Whh_h + 1024 * Hdim;              // 512*256

    // zero h0,h1,c0,c1,dinp (contiguous): 4*2048 + 4096 = 12288 floats
    {
        int nz = 4 * Bdim * Hdim + Bdim * OUTdim;
        init_zero_kernel<<<(nz + 255) / 256, 256, 0, stream>>>(h0, nz);
    }
    // pack all weights to f16 once
    pack_f16_kernel<<<(1024 * Ndim   + 255) / 256, 256, 0, stream>>>(enc_Wih, enc_Wih_h, 1024 * Ndim);
    pack_f16_kernel<<<(1024 * Hdim   + 255) / 256, 256, 0, stream>>>(enc_Whh, enc_Whh_h, 1024 * Hdim);
    pack_f16_kernel<<<(1024 * OUTdim + 255) / 256, 256, 0, stream>>>(dec_Wih, dec_Wih_h, 1024 * OUTdim);
    pack_f16_kernel<<<(1024 * Hdim   + 255) / 256, 256, 0, stream>>>(dec_Whh, dec_Whh_h, 1024 * Hdim);
    pack_f16_kernel<<<(OUTdim * Hdim + 255) / 256, 256, 0, stream>>>(fc_W, fc_W_h, OUTdim * Hdim);

    // GAT over all (b,t)
    gat_kernel<<<Bdim * Tdim, 256, 0, stream>>>(x, adj, gat_W, gat_a, xg);

    int cur = 0;  // which buffer currently holds (h, c)
    // encoder: x_gnn layout [B,T,N] -> time-slice pointer xg + t*N, row stride T*N
    for (int t = 0; t < Tdim; ++t) {
        float* hin  = cur ? h1 : h0;
        float* hout = cur ? h0 : h1;
        float* cin  = cur ? c1 : c0;
        float* cout = cur ? c0 : c1;
        lstm_step_kernel<<<Hdim / 16, 32, 0, stream>>>(
            xg + (long)t * Ndim, Ndim, Tdim * Ndim,
            enc_Wih_h, enc_Whh_h, enc_bih, enc_bhh, hin, cin, hout, cout);
        cur ^= 1;
    }
    // decoder with feedback
    for (int s = 0; s < FLEN; ++s) {
        float* hin  = cur ? h1 : h0;
        float* hout = cur ? h0 : h1;
        float* cin  = cur ? c1 : c0;
        float* cout = cur ? c0 : c1;
        lstm_step_kernel<<<Hdim / 16, 32, 0, stream>>>(
            dinp, OUTdim, OUTdim,
            dec_Wih_h, dec_Whh_h, dec_bih, dec_bhh, hin, cin, hout, cout);
        cur ^= 1;
        float* hnow = cur ? h1 : h0;
        fc_kernel<<<OUTdim / 16, 32, 0, stream>>>(hnow, fc_W_h, fc_b, dinp, out, s);
    }
}
